// AttentionLayer_17892833755530
// MI455X (gfx1250) — matrix operand order repaired and checked by reading, not compiled, not run on hardware
//
#include <hip/hip_runtime.h>
#include <hip/hip_bf16.h>

typedef __attribute__((ext_vector_type(16))) _Float16 v16h;
typedef __attribute__((ext_vector_type(8)))  _Float16 v8h;
typedef __attribute__((ext_vector_type(8)))  float    v8f;

constexpr int BSZ = 8, QN = 2048, KN = 2048, DIN = 1024, DPROJ = 1024;

constexpr int BM = 128, BN = 128, BK = 32;
constexpr int LDST = 40;          // LDS row stride in halves (padded, 16B-aligned rows)

enum { OUT_F16 = 0, OUT_F16_TRANS = 1, OUT_F32 = 2 };

// CDNA5 async global->LDS copy, 16B per lane. dst: LDS byte offset (group-
// relative; generic LDS pointers are aperture-form so low 32 bits work),
// src: per-lane global address. Tracked by ASYNCcnt.
__device__ __forceinline__ void async_g2l_b128(unsigned lds_off, const void* gaddr)
{
    asm volatile("global_load_async_to_lds_b128 %0, %1, off"
                 :: "v"(lds_off), "v"(gaddr) : "memory");
}
__device__ __forceinline__ void wait_asynccnt0()
{
    asm volatile("s_wait_asynccnt 0x0" ::: "memory");
}

// C[M,N] = A[M,K] * Bsrc[N,K]^T  (contraction over K), optional bias/alpha.
// A, Bsrc row-major. Batched via blockIdx.z with element strides.
// f16 operands: async-to-LDS issued before compute. fp32 operands: global
// loads issued before compute (registers), converted+ds_stored after compute
// so the loadcnt wait is hidden behind the WMMA block. Double-buffered LDS.
template <typename TA, typename TB, int MODE>
__global__ __launch_bounds__(256) void gemm_wmma(
    const TA* __restrict__ A, const TB* __restrict__ Bsrc,
    const float* __restrict__ bias, void* __restrict__ Out,
    int M, int N, int K, long aBatch, long bBatch, long oBatch, float alpha)
{
    __shared__ _Float16 lA[2][BM * LDST];
    __shared__ _Float16 lB[2][BN * LDST];

    const int tid    = threadIdx.x;
    const int wave   = tid >> 5;
    const int lane   = tid & 31;
    const int hi     = lane >> 4;        // 0: lanes 0-15, 1: lanes 16-31
    const int lanelo = lane & 15;
    const int wm     = wave & 1;         // 2 waves along M
    const int wn     = wave >> 1;        // 4 waves along N
    const int m0     = blockIdx.y * BM;
    const int n0     = blockIdx.x * BN;
    const int bz     = blockIdx.z;

    const TA* Ab = A    + (size_t)bz * aBatch;
    const TB* Bb = Bsrc + (size_t)bz * bBatch;

    constexpr bool ANY_ASYNC = (sizeof(TA) == 2) || (sizeof(TB) == 2);

    float4 ra[4], rb[4];                 // register staging for fp32 operands

    // Phase 1: issue all global traffic for tile at k0 (no waits here).
    auto stage_issue = [&](int buf, int k0) {
        if constexpr (sizeof(TA) == 4) {
#pragma unroll
            for (int i = 0; i < 4; ++i) {
                int lin = i * 256 + tid;
                int r = lin >> 3, c = (lin & 7) * 4;
                ra[i] = *(const float4*)(Ab + (size_t)(m0 + r) * K + k0 + c);
            }
        } else {
#pragma unroll
            for (int i = 0; i < 2; ++i) {
                int lin = i * 256 + tid;
                int r = lin >> 2, c = (lin & 3) * 8;
                const void* g = Ab + (size_t)(m0 + r) * K + k0 + c;
                async_g2l_b128((unsigned)(unsigned long long)&lA[buf][r * LDST + c], g);
            }
        }
        if constexpr (sizeof(TB) == 4) {
#pragma unroll
            for (int i = 0; i < 4; ++i) {
                int lin = i * 256 + tid;
                int r = lin >> 3, c = (lin & 7) * 4;
                rb[i] = *(const float4*)(Bb + (size_t)(n0 + r) * K + k0 + c);
            }
        } else {
#pragma unroll
            for (int i = 0; i < 2; ++i) {
                int lin = i * 256 + tid;
                int r = lin >> 2, c = (lin & 3) * 8;
                const void* g = Bb + (size_t)(n0 + r) * K + k0 + c;
                async_g2l_b128((unsigned)(unsigned long long)&lB[buf][r * LDST + c], g);
            }
        }
    };

    // Phase 2: convert registers and publish to LDS (fp32 operands only).
    auto stage_commit = [&](int buf) {
        if constexpr (sizeof(TA) == 4) {
#pragma unroll
            for (int i = 0; i < 4; ++i) {
                int lin = i * 256 + tid;
                int r = lin >> 3, c = (lin & 7) * 4;
                _Float16* d = &lA[buf][r * LDST + c];
                d[0] = (_Float16)ra[i].x; d[1] = (_Float16)ra[i].y;
                d[2] = (_Float16)ra[i].z; d[3] = (_Float16)ra[i].w;
            }
        }
        if constexpr (sizeof(TB) == 4) {
#pragma unroll
            for (int i = 0; i < 4; ++i) {
                int lin = i * 256 + tid;
                int r = lin >> 3, c = (lin & 7) * 4;
                _Float16* d = &lB[buf][r * LDST + c];
                d[0] = (_Float16)rb[i].x; d[1] = (_Float16)rb[i].y;
                d[2] = (_Float16)rb[i].z; d[3] = (_Float16)rb[i].w;
            }
        }
    };

    v8f acc[4][2];
#pragma unroll
    for (int mt = 0; mt < 4; ++mt)
#pragma unroll
        for (int nt = 0; nt < 2; ++nt) acc[mt][nt] = (v8f)(0.0f);

    // Prologue: tile 0 (first-tile latency is unavoidable).
    stage_issue(0, 0);
    stage_commit(0);

    const int steps = K / BK;
    for (int it = 0; it < steps; ++it) {
        const int cur = it & 1;
        if constexpr (ANY_ASYNC) wait_asynccnt0();   // own asyncs done pre-barrier
        __syncthreads();                             // all waves' tile 'cur' visible

        if (it + 1 < steps) stage_issue(cur ^ 1, (it + 1) * BK);

        // ---- fragments per ISA 16-bit A/B layouts ----
        v16h af[4], bf[2];
#pragma unroll
        for (int mt = 0; mt < 4; ++mt) {
            const _Float16* p = &lA[cur][(wm * 64 + mt * 16 + lanelo) * LDST + hi * 8];
            union { v16h v; v8h h[2]; } u;
            u.h[0] = *(const v8h*)p;          // K = kb .. kb+7   (kb = 8*hi)
            u.h[1] = *(const v8h*)(p + 16);   // K = kb+16 .. kb+23
            af[mt] = u.v;
        }
#pragma unroll
        for (int nt = 0; nt < 2; ++nt) {
            const _Float16* p = &lB[cur][(wn * 32 + nt * 16 + lanelo) * LDST + hi * 8];
            union { v16h v; v8h h[2]; } u;
            u.h[0] = *(const v8h*)p;          // K = hi*8 .. +7
            u.h[1] = *(const v8h*)(p + 16);   // K = hi*8+16 .. +23
            bf[nt] = u.v;
        }
#pragma unroll
        for (int mt = 0; mt < 4; ++mt)
#pragma unroll
            for (int nt = 0; nt < 2; ++nt)
                acc[mt][nt] = __builtin_amdgcn_wmma_f32_16x16x32_f16(
                    false, af[mt], false, bf[nt], (short)0, acc[mt][nt], false, false);

        // Convert + publish next tile after compute: loadcnt wait lands here,
        // hidden behind the 8 WMMAs above. ds_stores complete before the next
        // barrier (compiler emits s_wait_dscnt 0 before s_barrier_signal).
        if (it + 1 < steps) stage_commit(cur ^ 1);
    }

    // ---- epilogue per C/D VGPR layout: VGPR v -> M = v + 8*hi ----
#pragma unroll
    for (int mt = 0; mt < 4; ++mt) {
#pragma unroll
        for (int nt = 0; nt < 2; ++nt) {
            const int col   = n0 + wn * 32 + nt * 16 + lanelo;
            const int rbase = m0 + wm * 64 + mt * 16 + hi * 8;
            float bv = 0.0f;
            if constexpr (MODE != OUT_F32) bv = bias[col];
#pragma unroll
            for (int v = 0; v < 8; ++v) {
                const int row = rbase + v;
                const float val = acc[mt][nt][v];
                if constexpr (MODE == OUT_F32) {
                    ((float*)Out)[(size_t)bz * oBatch + (size_t)row * N + col] = val * alpha;
                } else if constexpr (MODE == OUT_F16) {
                    ((_Float16*)Out)[(size_t)row * N + col] = (_Float16)(val + bv);
                } else { // OUT_F16_TRANS: row = b*QN + k -> out[b][col][k]
                    const int bb = row >> 11, kk = row & 2047;
                    ((_Float16*)Out)[((size_t)bb * N + col) * (size_t)QN + kk] =
                        (_Float16)(val + bv);
                }
            }
        }
    }
}

// In-place masked softmax over one (b,q) row of KN scores; applies mask
// both pre-softmax (-1e9) and post-softmax (zero), matching the reference.
__global__ __launch_bounds__(256) void softmax_mask(float* __restrict__ attn,
                                                    const int* __restrict__ mask)
{
    const int row = blockIdx.x;        // b*QN + q
    const int b   = row >> 11;         // QN = 2048
    float* s = attn + (size_t)row * KN;
    const int* mrow = mask + (size_t)b * KN;

    __shared__ float red[256];
    float vals[8]; int mk[8];
    float mx = -1e30f;
#pragma unroll
    for (int i = 0; i < 8; ++i) {
        const int k = threadIdx.x + i * 256;
        mk[i] = mrow[k];
        float v = s[k];
        v = mk[i] ? v : -1.0e9f;
        vals[i] = v;
        mx = fmaxf(mx, v);
    }
    red[threadIdx.x] = mx; __syncthreads();
    for (int off = 128; off > 0; off >>= 1) {
        if (threadIdx.x < off) red[threadIdx.x] = fmaxf(red[threadIdx.x], red[threadIdx.x + off]);
        __syncthreads();
    }
    mx = red[0]; __syncthreads();

    float sum = 0.0f;
#pragma unroll
    for (int i = 0; i < 8; ++i) { vals[i] = __expf(vals[i] - mx); sum += vals[i]; }
    red[threadIdx.x] = sum; __syncthreads();
    for (int off = 128; off > 0; off >>= 1) {
        if (threadIdx.x < off) red[threadIdx.x] += red[threadIdx.x + off];
        __syncthreads();
    }
    const float inv = 1.0f / red[0];
#pragma unroll
    for (int i = 0; i < 8; ++i) {
        const int k = threadIdx.x + i * 256;
        s[k] = mk[i] ? vals[i] * inv : 0.0f;
    }
}

extern "C" void kernel_launch(void* const* d_in, const int* in_sizes, int n_in,
                              void* d_out, int out_size, void* d_ws, size_t ws_size,
                              hipStream_t stream)
{
    const float* query = (const float*)d_in[0];
    const float* key   = (const float*)d_in[1];
    const float* value = (const float*)d_in[2];
    const int*   mask  = (const int*)d_in[3];
    const float* Wq = (const float*)d_in[4]; const float* bq = (const float*)d_in[5];
    const float* Wk = (const float*)d_in[6]; const float* bk = (const float*)d_in[7];
    const float* Wv = (const float*)d_in[8]; const float* bv = (const float*)d_in[9];

    float* ctx  = (float*)d_out;
    float* attn = ctx + (size_t)BSZ * QN * DPROJ;

    _Float16* pq  = (_Float16*)d_ws;                      // [B*QN, DPROJ]
    _Float16* pk  = pq  + (size_t)BSZ * QN * DPROJ;       // [B*KN, DPROJ]
    _Float16* pvT = pk  + (size_t)BSZ * KN * DPROJ;       // [B, DPROJ, KN]

    const dim3 blk(256);

    // Projections: M = B*QN flattened, N = DPROJ, K = DIN
    {
        const dim3 g(DPROJ / BN, (BSZ * QN) / BM, 1);
        gemm_wmma<float, float, OUT_F16><<<g, blk, 0, stream>>>(
            query, Wq, bq, pq, BSZ * QN, DPROJ, DIN, 0, 0, 0, 1.0f);
        gemm_wmma<float, float, OUT_F16><<<g, blk, 0, stream>>>(
            key, Wk, bk, pk, BSZ * KN, DPROJ, DIN, 0, 0, 0, 1.0f);
        gemm_wmma<float, float, OUT_F16_TRANS><<<g, blk, 0, stream>>>(
            value, Wv, bv, pvT, BSZ * KN, DPROJ, DIN, 0, 0, 0, 1.0f);
    }

    // Scores: per-batch pq[2048,1024] x pk[2048,1024]^T * (1/32) -> attn fp32
    {
        const dim3 g(KN / BN, QN / BM, BSZ);
        gemm_wmma<_Float16, _Float16, OUT_F32><<<g, blk, 0, stream>>>(
            pq, pk, nullptr, attn, QN, KN, DPROJ,
            (long)QN * DPROJ, (long)KN * DPROJ, (long)QN * KN, 0.03125f);
    }

    // Masked softmax in place on attn
    softmax_mask<<<BSZ * QN, blk, 0, stream>>>(attn, mask);

    // Context: per-batch attn[2048,2048] (fp32->f16 staged) x pvT[1024,2048]^T -> ctx
    {
        const dim3 g(DPROJ / BN, QN / BM, BSZ);
        gemm_wmma<float, _Float16, OUT_F32><<<g, blk, 0, stream>>>(
            attn, pvT, nullptr, ctx, QN, DPROJ, KN,
            (long)QN * KN, (long)DPROJ * KN, (long)QN * DPROJ, 1.0f);
    }
}
